// HeteGAT_multi_RL2_36223754174748
// MI455X (gfx1250) — matrix operand order repaired and checked by reading, not compiled, not run on hardware
//
#include <hip/hip_runtime.h>
#include <hip/hip_bf16.h>
#include <stdint.h>

// Problem sizes (from reference)
#define B_N    2048
#define FEAT   302
#define FEATP  320      // padded K for first GEMM (mult of 32)
#define HID    256
#define NBN    6000
#define NI     5000
#define NIP    5120     // padded to mult of 128 (full 8-wave GEMM blocks, barrier-safe)
#define NE     65536
#define RNUM   3
#define NHEAD  4
#define HDIM   64

typedef __attribute__((ext_vector_type(16))) __bf16 v16bf;
typedef __attribute__((ext_vector_type(8)))  float  v8f;

static __device__ __forceinline__ uint16_t f2bf(float f) {
    uint32_t u = __float_as_uint(f);
    uint32_t r = u + 0x7FFFu + ((u >> 16) & 1u);   // round-to-nearest-even
    return (uint16_t)(r >> 16);
}

struct Frag32 { uint4 lo, hi; };   // 32 bytes == v16bf

// gfx1250 async copy global->LDS (ASYNCcnt-tracked), per CDNA5 ISA
static __device__ __forceinline__ void async_load_b128(uint32_t lds_off, const void* gptr) {
    asm volatile("global_load_async_to_lds_b128 %0, %1, off"
                 :: "v"(lds_off), "v"((uint64_t)(uintptr_t)gptr)
                 : "memory");
}
static __device__ __forceinline__ void wait_asynccnt0() {
    asm volatile("s_wait_asynccnt 0x0" ::: "memory");
}

// ---------------------------------------------------------------------------
// bf16 WMMA GEMM:  C[M,N] (f32) = A[M,K] * Bt[N,K]^T (+ bias[N])
// Block = 8 waves = 128x64 output tile. Each wave owns 16 rows x 64 cols
// (4 accumulators; A fragment reused 4x). The 64x32 B panel for the current
// k-step is shared by all 8 waves: staged into LDS with
// GLOBAL_LOAD_ASYNC_TO_LDS_B128 (double buffered, s_wait_asynccnt + barrier),
// then read back as WMMA fragments with ds_load_b128.
// Requires: M % 128 == 0, N % 64 == 0, K % 32 == 0, 16B-aligned buffers.
// Fragment layouts per CDNA5 ISA 7.12.2:
//   A 16x32: lane<16 -> row M=lane, K {0..7,16..23}; lane>=16 -> K {8..15,24..31}
//   B 32x16: lane<16 -> col N=lane, K {0..7,16..23}; lane>=16 -> K {8..15,24..31}
//   C/D:     lane%16 = N, vgpr r -> M = r + 8*(lane/16)
// ---------------------------------------------------------------------------
__global__ __launch_bounds__(256)
void k_gemm_bf16(const uint16_t* __restrict__ A, int lda,
                 const uint16_t* __restrict__ Bt, int ldb,
                 float* __restrict__ C, int ldc,
                 int M, int N, int K, const float* __restrict__ bias)
{
    __shared__ __align__(16) uint16_t bs[2][64 * 32];   // 2 x 4KB double buffer
    int wave = threadIdx.x >> 5;
    int lane = threadIdx.x & 31;
    int mb = M >> 7;                        // 128-row blocks
    int m0 = (blockIdx.x % mb) * 128 + wave * 16;
    int n0 = (blockIdx.x / mb) * 64;
    int half = lane >> 4;
    int l16  = lane & 15;

    const uint16_t* arow = A + (size_t)(m0 + l16) * lda + half * 8;

    // async staging map: thread t copies 16B; 64 rows x 64B panel = 256 x 16B
    int srow = threadIdx.x >> 2;            // 0..63 (panel row = Bt row n0+srow)
    int sseg = (threadIdx.x & 3) * 8;       // element offset within 32-elem k-slab
    const uint16_t* gsrc = Bt + (size_t)(n0 + srow) * ldb + sseg;
    uint32_t lds0 = (uint32_t)(uintptr_t)&bs[0][srow * 32 + sseg];
    uint32_t lds1 = (uint32_t)(uintptr_t)&bs[1][srow * 32 + sseg];

    int nsteps = K >> 5;
    async_load_b128(lds0, gsrc);            // prologue: stage k-step 0

    v8f acc[4] = {};
    for (int s = 0; s < nsteps; ++s) {
        int kk = s << 5;
        int cur = s & 1;
        wait_asynccnt0();                   // my async copies landed in LDS
        __syncthreads();                    // everyone's copies landed
        if (s + 1 < nsteps)                 // prefetch next k-slab into other buffer
            async_load_b128(cur ? lds0 : lds1, gsrc + kk + 32);

        Frag32 fa;
        fa.lo = *reinterpret_cast<const uint4*>(arow + kk);        // K {0..7}/{8..15}
        fa.hi = *reinterpret_cast<const uint4*>(arow + kk + 16);   // K {16..23}/{24..31}
        v16bf a = __builtin_bit_cast(v16bf, fa);

        const uint16_t* bbase = &bs[cur][0];
#pragma unroll
        for (int j = 0; j < 4; ++j) {
            const uint16_t* bp = bbase + (j * 16 + l16) * 32 + half * 8;
            Frag32 fb;
            fb.lo = *reinterpret_cast<const uint4*>(bp);
            fb.hi = *reinterpret_cast<const uint4*>(bp + 16);
            v16bf b = __builtin_bit_cast(v16bf, fb);
            acc[j] = __builtin_amdgcn_wmma_f32_16x16x32_bf16(false, a, false, b,
                                                             (short)0, acc[j], false, false);
        }
        __syncthreads();                    // done reading cur before it is rewritten
    }

    int crow = m0 + half * 8;
#pragma unroll
    for (int j = 0; j < 4; ++j) {
        int ccol = n0 + j * 16 + l16;
        float bv = bias ? bias[ccol] : 0.0f;
#pragma unroll
        for (int rr = 0; rr < 8; ++rr)
            C[(size_t)(crow + rr) * ldc + ccol] = acc[j][rr] + bv;
    }
}

// ---------------------------------------------------------------------------
// Small helper kernels
// ---------------------------------------------------------------------------
__global__ void k_zero_f32(float* p, int n) {
    int i = blockIdx.x * blockDim.x + threadIdx.x;
    if (i < n) p[i] = 0.0f;
}
__global__ void k_zero_u32(unsigned* p, int n) {
    int i = blockIdx.x * blockDim.x + threadIdx.x;
    if (i < n) p[i] = 0u;
}
// W[K,N] f32  ->  Wt[N,Kp] bf16 (zero-padded K)
__global__ void k_wt_bf16(const float* __restrict__ W, uint16_t* __restrict__ Wt,
                          int K, int N, int Kp)
{
    int i = blockIdx.x * blockDim.x + threadIdx.x;
    if (i >= N * Kp) return;
    int n = i / Kp, k = i % Kp;
    Wt[i] = (k < K) ? f2bf(W[(size_t)k * N + n]) : (uint16_t)0;
}
__global__ void k_f32_to_bf16(const float* __restrict__ X, uint16_t* __restrict__ Y, int n) {
    int i = blockIdx.x * blockDim.x + threadIdx.x;
    if (i < n) Y[i] = f2bf(X[i]);
}
__global__ void k_elu_bf16(const float* __restrict__ X, uint16_t* __restrict__ Y, int n) {
    int i = blockIdx.x * blockDim.x + threadIdx.x;
    if (i >= n) return;
    float v = X[i];
    v = v > 0.0f ? v : (__expf(v) - 1.0f);
    Y[i] = f2bf(v);
}
// features[n_ids] -> padded bf16 [NIP, FEATP]
__global__ void k_gather_feat(const float* __restrict__ F, const int* __restrict__ ids,
                              uint16_t* __restrict__ Y)
{
    int row = blockIdx.x;
    for (int k = threadIdx.x; k < FEATP; k += blockDim.x) {
        float v = 0.0f;
        if (row < NI && k < FEAT) v = F[(size_t)ids[row] * FEAT + k];
        Y[(size_t)row * FEATP + k] = f2bf(v);
    }
}
__global__ void k_time(const float* __restrict__ F, const int* __restrict__ bn,
                       float* __restrict__ t)
{
    int b = blockIdx.x * blockDim.x + threadIdx.x;
    if (b < B_N) t[b] = F[(size_t)bn[b] * FEAT + (FEAT - 2)] * 10000.0f;
}
// BN stats over the real M rows of X[Mp,256]: scale/shift per column
__global__ void k_bn_stats(const float* __restrict__ X, int Mreal,
                           const float* __restrict__ gamma, const float* __restrict__ beta,
                           float* __restrict__ scale, float* __restrict__ shift)
{
    __shared__ float sh[256], sh2[256];
    int c = blockIdx.x;
    float s = 0.0f, s2 = 0.0f;
    for (int r = threadIdx.x; r < Mreal; r += blockDim.x) {
        float v = X[(size_t)r * HID + c];
        s += v; s2 += v * v;
    }
    sh[threadIdx.x] = s; sh2[threadIdx.x] = s2; __syncthreads();
    for (int o = 128; o > 0; o >>= 1) {
        if (threadIdx.x < o) { sh[threadIdx.x] += sh[threadIdx.x + o];
                               sh2[threadIdx.x] += sh2[threadIdx.x + o]; }
        __syncthreads();
    }
    if (threadIdx.x == 0) {
        float mu  = sh[0] / Mreal;
        float var = sh2[0] / Mreal - mu * mu;   // biased, matches torch BN
        float sc  = gamma[c] * rsqrtf(var + 1e-5f);
        scale[c] = sc; shift[c] = beta[c] - mu * sc;
    }
}
__global__ void k_bn_elu_bf16(const float* __restrict__ X, const float* __restrict__ scale,
                              const float* __restrict__ shift, uint16_t* __restrict__ Y, int n)
{
    int i = blockIdx.x * blockDim.x + threadIdx.x;
    if (i >= n) return;
    int c = i & (HID - 1);
    float v = X[i] * scale[c] + shift[c];
    v = v > 0.0f ? v : (__expf(v) - 1.0f);
    Y[i] = f2bf(v);
}
// dot(row of X[.,256], w) per warp
__global__ __launch_bounds__(256)
void k_dot256(const float* __restrict__ X, const float* __restrict__ w,
              float* __restrict__ out, int rows)
{
    int warp = (blockIdx.x * blockDim.x + threadIdx.x) >> 5;
    int lane = threadIdx.x & 31;
    if (warp >= rows) return;
    const float* row = X + (size_t)warp * HID;
    float s = 0.0f;
    for (int k = lane; k < HID; k += 32) s += row[k] * w[k];
    for (int o = 16; o > 0; o >>= 1) s += __shfl_down(s, o, 32);
    if (lane == 0) out[warp] = s;
}
// f1/f2: out[h*B+b] = dot(fts[h,b,:], w[h]) + bias[h]
__global__ __launch_bounds__(256)
void k_head_dot64(const float* __restrict__ fts, const float* __restrict__ w,
                  const float* __restrict__ bias, float* __restrict__ out)
{
    int warp = (blockIdx.x * blockDim.x + threadIdx.x) >> 5;
    int lane = threadIdx.x & 31;
    if (warp >= NHEAD * B_N) return;
    int h = warp / B_N;
    const float* row = fts + (size_t)warp * HDIM;
    float s = 0.0f;
    for (int k = lane; k < HDIM; k += 32) s += row[k] * w[h * HDIM + k];
    for (int o = 16; o > 0; o >>= 1) s += __shfl_down(s, o, 32);
    if (lane == 0) out[warp] = s + bias[h];
}
// fts[h][b][o] f32 -> ftsT[h][o][b] bf16  (V^T for the P@V WMMA GEMM)
__global__ void k_ftsT(const float* __restrict__ fts, uint16_t* __restrict__ ftsT) {
    int i = blockIdx.x * blockDim.x + threadIdx.x;
    if (i >= NHEAD * B_N * HDIM) return;
    int h = i / (B_N * HDIM), rem = i % (B_N * HDIM);
    int b = rem / HDIM, o = rem % HDIM;
    ftsT[((size_t)h * HDIM + o) * B_N + b] = f2bf(fts[i]);
}

// --- edge segment-softmax (GAT) ---
static __device__ __forceinline__ unsigned fkey(float f) {
    unsigned u = __float_as_uint(f);
    return (u & 0x80000000u) ? ~u : (u | 0x80000000u);
}
static __device__ __forceinline__ float fkeyinv(unsigned k) {
    unsigned u = (k & 0x80000000u) ? (k & 0x7FFFFFFFu) : ~k;
    return __uint_as_float(u);
}
__global__ void k_edge1(const int* __restrict__ src, const int* __restrict__ dst,
                        const float* __restrict__ asrc, const float* __restrict__ adst,
                        float* __restrict__ e, unsigned* __restrict__ segmax)
{
    int i = blockIdx.x * blockDim.x + threadIdx.x;
    if (i >= NE) return;
    float v = asrc[src[i]] + adst[dst[i]];
    v = v > 0.0f ? v : 0.2f * v;               // leaky_relu 0.2
    e[i] = v;
    atomicMax(&segmax[dst[i]], fkey(v));
}
__global__ void k_edge2(const int* __restrict__ dst, const float* __restrict__ e,
                        const unsigned* __restrict__ segmax,
                        float* __restrict__ ex, float* __restrict__ segsum)
{
    int i = blockIdx.x * blockDim.x + threadIdx.x;
    if (i >= NE) return;
    float v = __expf(e[i] - fkeyinv(segmax[dst[i]]));
    ex[i] = v;
    atomicAdd(&segsum[dst[i]], v);
}
__global__ __launch_bounds__(256)
void k_edge_scatter(const int* __restrict__ src, const int* __restrict__ dst,
                    const float* __restrict__ ex, const float* __restrict__ segsum,
                    const float* __restrict__ hsrc, float* __restrict__ fe)
{
    int eidx = blockIdx.x * 4 + (threadIdx.x >> 6);   // 4 edges / block, 64 thr/edge
    int f4   = (threadIdx.x & 63) * 4;
    if (eidx >= NE) return;
    int d = dst[eidx];
    float w = ex[eidx] / segsum[d];
    const float* hs = hsrc + (size_t)src[eidx] * HID + f4;
    float* fo = fe + (size_t)d * HID + f4;
    atomicAdd(fo + 0, w * hs[0]);
    atomicAdd(fo + 1, w * hs[1]);
    atomicAdd(fo + 2, w * hs[2]);
    atomicAdd(fo + 3, w * hs[3]);
}
__global__ void k_fe_fin(float* __restrict__ fe, const float* __restrict__ gbias,
                         uint16_t* __restrict__ febf)
{
    int i = blockIdx.x * blockDim.x + threadIdx.x;
    if (i >= B_N * HID) return;
    float v = fe[i] + gbias[i & (HID - 1)];
    fe[i] = v;
    febf[i] = f2bf(v);
}
// pre[b][n] = bias_r[bn[b], bn[n]] - |t[b]-t[n]|/5   (shared across heads)
__global__ __launch_bounds__(256)
void k_pre(const float* __restrict__ bias, const int* __restrict__ bn,
           const float* __restrict__ t, float* __restrict__ pre)
{
    int b = blockIdx.x;
    const float* brow = bias + (size_t)bn[b] * NBN;
    float tb = t[b];
    float* prow = pre + (size_t)b * B_N;
    for (int n = threadIdx.x; n < B_N; n += blockDim.x)
        prow[n] = brow[bn[n]] - fabsf(tb - t[n]) * 0.2f;
}
// row softmax of leaky_relu(f1[b]+f2[n]) + pre[b][n] -> P (bf16) for head h
__global__ __launch_bounds__(256)
void k_softmax_p(const float* __restrict__ f1, const float* __restrict__ f2,
                 const float* __restrict__ pre, uint16_t* __restrict__ P, int h)
{
    __shared__ float srow[B_N];
    __shared__ float red[256];
    int b = blockIdx.x;
    float f1b = f1[h * B_N + b];
    const float* f2h  = f2  + h * B_N;
    const float* prow = pre + (size_t)b * B_N;
    float mx = -3.4e38f;
    for (int n = threadIdx.x; n < B_N; n += 256) {
        float x = f1b + f2h[n];
        x = x > 0.0f ? x : 0.2f * x;
        x += prow[n];
        srow[n] = x;
        mx = fmaxf(mx, x);
    }
    red[threadIdx.x] = mx; __syncthreads();
    for (int o = 128; o > 0; o >>= 1) {
        if (threadIdx.x < o) red[threadIdx.x] = fmaxf(red[threadIdx.x], red[threadIdx.x + o]);
        __syncthreads();
    }
    mx = red[0]; __syncthreads();
    float sm = 0.0f;
    for (int n = threadIdx.x; n < B_N; n += 256) {
        float v = __expf(srow[n] - mx);
        srow[n] = v; sm += v;
    }
    red[threadIdx.x] = sm; __syncthreads();
    for (int o = 128; o > 0; o >>= 1) {
        if (threadIdx.x < o) red[threadIdx.x] += red[threadIdx.x + o];
        __syncthreads();
    }
    float inv = 1.0f / red[0];
    uint16_t* prowP = P + (size_t)b * B_N;
    for (int n = threadIdx.x; n < B_N; n += 256) prowP[n] = f2bf(srow[n] * inv);
}
// final: alphas = softmax_r(dot(tanh(vpre[b,r,:]), u)); out = sum_r multi[b,r,:]*alpha[r]
__global__ __launch_bounds__(256)
void k_final(const float* __restrict__ vpre, const float* __restrict__ uo,
             const float* __restrict__ multi, float* __restrict__ out)
{
    __shared__ float red[256];
    __shared__ float dots[RNUM];
    int b = blockIdx.x;
    for (int r = 0; r < RNUM; ++r) {
        float v = tanhf(vpre[((size_t)b * RNUM + r) * HID + threadIdx.x]) * uo[threadIdx.x];
        red[threadIdx.x] = v; __syncthreads();
        for (int o = 128; o > 0; o >>= 1) {
            if (threadIdx.x < o) red[threadIdx.x] += red[threadIdx.x + o];
            __syncthreads();
        }
        if (threadIdx.x == 0) dots[r] = red[0];
        __syncthreads();
    }
    float m  = fmaxf(dots[0], fmaxf(dots[1], dots[2]));
    float e0 = __expf(dots[0] - m), e1 = __expf(dots[1] - m), e2 = __expf(dots[2] - m);
    float inv = 1.0f / (e0 + e1 + e2);
    size_t base = (size_t)b * RNUM * HID;
    out[(size_t)b * HID + threadIdx.x] =
        e0 * inv * multi[base + threadIdx.x] +
        e1 * inv * multi[base + HID + threadIdx.x] +
        e2 * inv * multi[base + 2 * HID + threadIdx.x];
}

// ---------------------------------------------------------------------------
extern "C" void kernel_launch(void* const* d_in, const int* in_sizes, int n_in,
                              void* d_out, int out_size, void* d_ws, size_t ws_size,
                              hipStream_t stream)
{
    (void)in_sizes; (void)n_in; (void)out_size; (void)ws_size;
    const float* features = (const float*)d_in[0];
    const float* biases[RNUM] = {(const float*)d_in[1], (const float*)d_in[2], (const float*)d_in[3]};
    const int*   batch_nodes  = (const int*)d_in[4];
    const int*   n_ids[RNUM]  = {(const int*)d_in[5], (const int*)d_in[6], (const int*)d_in[7]};
    const int*   eidx[RNUM]   = {(const int*)d_in[8], (const int*)d_in[9], (const int*)d_in[10]};
    // d_in[11] = RL_thresholds (unused by reference)
    const float* mlp_w1   = (const float*)d_in[12];
    const float* mlp_b1   = (const float*)d_in[13];
    const float* bn_gamma = (const float*)d_in[14];
    const float* bn_beta  = (const float*)d_in[15];
    const float* mlp_w2   = (const float*)d_in[16];
    const float* mlp_b2   = (const float*)d_in[17];
    // 18..21: hop_att0/hop_att1/hop_bias0/hop_bias1 -> dead code in reference
    const float* gat_w    = (const float*)d_in[22];
    const float* gat_asrc = (const float*)d_in[23];
    const float* gat_adst = (const float*)d_in[24];
    const float* gat_bias = (const float*)d_in[25];
    const float* t_w      = (const float*)d_in[26];
    const float* t_b      = (const float*)d_in[27];
    const float* t_f1_w   = (const float*)d_in[28];
    const float* t_f1_b   = (const float*)d_in[29];
    const float* t_f2_w   = (const float*)d_in[30];
    const float* t_f2_b   = (const float*)d_in[31];
    const float* fin_w    = (const float*)d_in[32];
    const float* fin_b    = (const float*)d_in[33];
    const float* w_omega  = (const float*)d_in[34];
    const float* b_omega  = (const float*)d_in[35];
    const float* u_omega  = (const float*)d_in[36];
    float* out = (float*)d_out;

    // ---- workspace carve (256B aligned) ----
    char* ws = (char*)d_ws; size_t off = 0;
    auto alloc = [&](size_t bytes) -> void* {
        void* p = ws + off; off += (bytes + 255) & ~(size_t)255; return p;
    };
    uint16_t* W1t    = (uint16_t*)alloc((size_t)HID * FEATP * 2);
    uint16_t* W2t    = (uint16_t*)alloc((size_t)HID * HID * 2);
    uint16_t* gatWt  = (uint16_t*)alloc((size_t)RNUM * HID * HID * 2);
    uint16_t* twT    = (uint16_t*)alloc((size_t)RNUM * NHEAD * HDIM * HDIM * 2);
    uint16_t* finWt  = (uint16_t*)alloc((size_t)HID * HID * 2);
    uint16_t* womT   = (uint16_t*)alloc((size_t)HID * HID * 2);
    float* bn_scale  = (float*)alloc(HID * 4);
    float* bn_shift  = (float*)alloc(HID * 4);
    float* btime     = (float*)alloc(B_N * 4);
    uint16_t* xfeat  = (uint16_t*)alloc((size_t)NIP * FEATP * 2);
    float* h1        = (float*)alloc((size_t)NIP * HID * 4);     // also reused as xm f32
    uint16_t* act1   = (uint16_t*)alloc((size_t)NIP * HID * 2);
    uint16_t* xmbf   = (uint16_t*)alloc((size_t)NIP * HID * 2);
    float* hsrc      = (float*)alloc((size_t)NIP * HID * 4);
    float* asrc      = (float*)alloc(NIP * 4);
    float* adst      = (float*)alloc(B_N * 4);
    float* ebuf      = (float*)alloc(NE * 4);
    float* exbuf     = (float*)alloc(NE * 4);
    unsigned* segmax = (unsigned*)alloc(B_N * 4);
    float* segsum    = (float*)alloc(B_N * 4);
    float* fe        = (float*)alloc((size_t)B_N * HID * 4);
    uint16_t* febf   = (uint16_t*)alloc((size_t)B_N * HID * 2);
    float* fts       = (float*)alloc((size_t)NHEAD * B_N * HDIM * 4);
    uint16_t* ftsT   = (uint16_t*)alloc((size_t)NHEAD * HDIM * B_N * 2);
    float* f1        = (float*)alloc(NHEAD * B_N * 4);
    float* f2        = (float*)alloc(NHEAD * B_N * 4);
    float* pre       = (float*)alloc((size_t)B_N * B_N * 4);
    uint16_t* Pm     = (uint16_t*)alloc((size_t)B_N * B_N * 2);
    float* h2        = (float*)alloc((size_t)B_N * HID * 4);
    uint16_t* h2bf   = (uint16_t*)alloc((size_t)B_N * HID * 2);
    float* multi     = (float*)alloc((size_t)B_N * RNUM * HID * 4);
    uint16_t* multibf= (uint16_t*)alloc((size_t)B_N * RNUM * HID * 2);
    float* vpre      = (float*)alloc((size_t)B_N * RNUM * HID * 4);

    auto gemm = [&](const uint16_t* A, int lda, const uint16_t* Bt, int ldb,
                    float* C, int ldc, int M, int N, int K, const float* bias) {
        int blocks = (M / 128) * (N / 64);
        k_gemm_bf16<<<blocks, 256, 0, stream>>>(A, lda, Bt, ldb, C, ldc, M, N, K, bias);
    };
    auto ew = [&](int n) { return dim3((n + 255) / 256); };

    // ---- weight prep (bf16, transposed) ----
    k_wt_bf16<<<ew(HID * FEATP), 256, 0, stream>>>(mlp_w1, W1t, FEAT, HID, FEATP);
    k_wt_bf16<<<ew(HID * HID), 256, 0, stream>>>(mlp_w2, W2t, HID, HID, HID);
    for (int r = 0; r < RNUM; ++r)
        k_wt_bf16<<<ew(HID * HID), 256, 0, stream>>>(gat_w + (size_t)r * HID * HID,
                                                     gatWt + (size_t)r * HID * HID, HID, HID, HID);
    for (int rh = 0; rh < RNUM * NHEAD; ++rh)
        k_wt_bf16<<<ew(HDIM * HDIM), 256, 0, stream>>>(t_w + (size_t)rh * HDIM * HDIM,
                                                       twT + (size_t)rh * HDIM * HDIM, HDIM, HDIM, HDIM);
    k_wt_bf16<<<ew(HID * HID), 256, 0, stream>>>(fin_w, finWt, HID, HID, HID);
    k_wt_bf16<<<ew(HID * HID), 256, 0, stream>>>(w_omega, womT, HID, HID, HID);
    k_time<<<ew(B_N), 256, 0, stream>>>(features, batch_nodes, btime);

    for (int r = 0; r < RNUM; ++r) {
        const int* esrc = eidx[r];
        const int* edst = eidx[r] + NE;
        // xm = _mlp(features[n_ids_r])
        k_gather_feat<<<NIP, 256, 0, stream>>>(features, n_ids[r], xfeat);
        gemm(xfeat, FEATP, W1t, FEATP, h1, HID, NIP, HID, FEATP, mlp_b1);
        k_bn_stats<<<HID, 256, 0, stream>>>(h1, NI, bn_gamma, bn_beta, bn_scale, bn_shift);
        k_bn_elu_bf16<<<ew(NIP * HID), 256, 0, stream>>>(h1, bn_scale, bn_shift, act1, NIP * HID);
        gemm(act1, HID, W2t, HID, h1, HID, NIP, HID, HID, mlp_b2);      // h1 := xm
        k_f32_to_bf16<<<ew(NIP * HID), 256, 0, stream>>>(h1, xmbf, NIP * HID);
        // GAT conv: h_src = xm @ W_r ; h_dst = h_src[:B]
        gemm(xmbf, HID, gatWt + (size_t)r * HID * HID, HID, hsrc, HID, NIP, HID, HID, nullptr);
        k_dot256<<<ew(NI * 32), 256, 0, stream>>>(hsrc, gat_asrc + r * HID, asrc, NI);
        k_dot256<<<ew(B_N * 32), 256, 0, stream>>>(hsrc, gat_adst + r * HID, adst, B_N);
        k_zero_u32<<<ew(B_N), 256, 0, stream>>>(segmax, B_N);
        k_zero_f32<<<ew(B_N), 256, 0, stream>>>(segsum, B_N);
        k_zero_f32<<<ew(B_N * HID), 256, 0, stream>>>(fe, B_N * HID);
        k_edge1<<<ew(NE), 256, 0, stream>>>(esrc, edst, asrc, adst, ebuf, segmax);
        k_edge2<<<ew(NE), 256, 0, stream>>>(edst, ebuf, segmax, exbuf, segsum);
        k_edge_scatter<<<NE / 4, 256, 0, stream>>>(esrc, edst, exbuf, segsum, hsrc, fe);
        k_fe_fin<<<ew(B_N * HID), 256, 0, stream>>>(fe, gat_bias + r * HID, febf);
        // temporal heads: fts = fe_h @ t_w[r,h] + t_b[r,h]
        for (int h = 0; h < NHEAD; ++h)
            gemm(febf + h * HDIM, HID, twT + (size_t)(r * NHEAD + h) * HDIM * HDIM, HDIM,
                 fts + (size_t)h * B_N * HDIM, HDIM, B_N, HDIM, HDIM,
                 t_b + (size_t)(r * NHEAD + h) * HDIM);
        k_head_dot64<<<ew(NHEAD * B_N * 32), 256, 0, stream>>>(fts, t_f1_w + r * NHEAD * HDIM,
                                                               t_f1_b + r * NHEAD, f1);
        k_head_dot64<<<ew(NHEAD * B_N * 32), 256, 0, stream>>>(fts, t_f2_w + r * NHEAD * HDIM,
                                                               t_f2_b + r * NHEAD, f2);
        k_ftsT<<<ew(NHEAD * B_N * HDIM), 256, 0, stream>>>(fts, ftsT);
        k_pre<<<B_N, 256, 0, stream>>>(biases[r], batch_nodes, btime, pre);
        // per head: P = softmax(...) ; h2[:, h*64:] = P @ fts[h]
        for (int h = 0; h < NHEAD; ++h) {
            k_softmax_p<<<B_N, 256, 0, stream>>>(f1, f2, pre, Pm, h);
            gemm(Pm, B_N, ftsT + (size_t)h * HDIM * B_N, B_N,
                 h2 + h * HDIM, HID, B_N, HDIM, B_N, nullptr);
        }
        k_elu_bf16<<<ew(B_N * HID), 256, 0, stream>>>(h2, h2bf, B_N * HID);
        // embeds[r] = h2 @ fin_w + fin_b, written into multi[:, r, :]
        gemm(h2bf, HID, finWt, HID, multi + r * HID, RNUM * HID, B_N, HID, HID, fin_b);
    }
    // final attention over R
    k_f32_to_bf16<<<ew(B_N * RNUM * HID), 256, 0, stream>>>(multi, multibf, B_N * RNUM * HID);
    gemm(multibf, HID, womT, HID, vpre, HID, B_N * RNUM, HID, HID, b_omega);
    k_final<<<B_N, 256, 0, stream>>>(vpre, u_omega, multi, out);
}
